// DFF_16887811408425
// MI455X (gfx1250) — matrix operand order repaired and checked by reading, not compiled, not run on hardware
//
#include <hip/hip_runtime.h>

// ---------------------------------------------------------------------------
// Problem constants (match reference)
// ---------------------------------------------------------------------------
constexpr int BATCH = 4;
constexpr int CIN   = 64;      // channels per source tensor
constexpr int HH    = 96;
constexpr int WW    = 96;
constexpr int HWSZ  = HH * WW;         // 9216
constexpr int OUT   = 64;              // output channels of deform conv
constexpr int KP    = 9;               // 3x3 positions
constexpr int OFFC  = 2 * KP;          // 18 offset channels
constexpr int CIN3  = 3 * CIN;         // 192 concat channels
constexpr int KTOT  = CIN3 * KP;       // 1728 = GEMM K
constexpr int KC    = 32;              // K per bf16 WMMA
constexpr int NCH   = KTOT / KC;       // 54 chunks
constexpr int MT    = 128;             // pixels per block (8 waves x 16)

constexpr size_t FEAT_ELEMS = (size_t)BATCH * OUT * HWSZ;   // 2,359,296
constexpr size_t OFF_ELEMS  = (size_t)BATCH * OFFC * HWSZ;  // 663,552
constexpr size_t WB_ELEMS   = (size_t)OUT * KTOT;           // 110,592

typedef __bf16 bf16_t;
typedef __attribute__((ext_vector_type(16))) __bf16 v16bf;
typedef __attribute__((ext_vector_type(8)))  float  v8f;
typedef int v4i __attribute__((vector_size(16)));   // matches builtin param type

union V16 { uint4 q[2]; v16bf v; };

// CDNA5 async global->LDS path (ASYNCcnt-tracked), with safe fallback.
#if defined(__has_builtin)
#if __has_builtin(__builtin_amdgcn_global_load_async_to_lds_b128)
#define HAVE_ASYNC_LDS 1
#endif
#endif
#define GAS __attribute__((address_space(1)))
#define LAS __attribute__((address_space(3)))

// ---------------------------------------------------------------------------
// Kernel 1: diff = (ref - dist)^2   (elementwise, float4 vectorized)
// ---------------------------------------------------------------------------
__global__ void diff_kernel(const float* __restrict__ ref,
                            const float* __restrict__ dist,
                            float* __restrict__ diff) {
    int i = blockIdx.x * blockDim.x + threadIdx.x;           // float4 index
    const float4* r4 = reinterpret_cast<const float4*>(ref);
    const float4* d4 = reinterpret_cast<const float4*>(dist);
    float4*       o4 = reinterpret_cast<float4*>(diff);
    float4 r = r4[i], d = d4[i];
    float4 o;
    o.x = (r.x - d.x) * (r.x - d.x);
    o.y = (r.y - d.y) * (r.y - d.y);
    o.z = (r.z - d.z) * (r.z - d.z);
    o.w = (r.w - d.w) * (r.w - d.w);
    o4[i] = o;
}

// ---------------------------------------------------------------------------
// Kernel 2: offset = conv3x3(ref, offset_w) + offset_b  -> ws (f32)
// Block: 256 threads = one 16x16 pixel tile, all 18 output channels.
// ---------------------------------------------------------------------------
__global__ void offset_conv_kernel(const float* __restrict__ x,
                                   const float* __restrict__ w,
                                   const float* __restrict__ bias,
                                   float* __restrict__ off) {
    __shared__ float tile[18 * 18];
    const int tid = threadIdx.x;
    const int ty = tid >> 4, tx = tid & 15;
    const int b  = blockIdx.z;
    const int h0 = blockIdx.y * 16, w0 = blockIdx.x * 16;
    const int h = h0 + ty, ww_ = w0 + tx;

    float acc[OFFC];
#pragma unroll
    for (int oc = 0; oc < OFFC; ++oc) acc[oc] = 0.f;

    for (int c = 0; c < CIN; ++c) {
        const float* xc = x + ((size_t)b * CIN + c) * HWSZ;
        for (int i = tid; i < 324; i += 256) {
            int r = i / 18, cc = i % 18;
            int gh = h0 - 1 + r, gw = w0 - 1 + cc;
            bool ok = (gh >= 0) & (gh < HH) & (gw >= 0) & (gw < WW);
            tile[i] = ok ? xc[gh * WW + gw] : 0.f;
        }
        __syncthreads();

        float in9[9];
#pragma unroll
        for (int ky = 0; ky < 3; ++ky)
#pragma unroll
            for (int kx = 0; kx < 3; ++kx)
                in9[ky * 3 + kx] = tile[(ty + ky) * 18 + (tx + kx)];

#pragma unroll
        for (int oc = 0; oc < OFFC; ++oc) {
            const float* wp = w + ((size_t)oc * CIN + c) * 9;   // uniform -> SGPR
#pragma unroll
            for (int k = 0; k < 9; ++k) acc[oc] = fmaf(wp[k], in9[k], acc[oc]);
        }
        __syncthreads();
    }

#pragma unroll
    for (int oc = 0; oc < OFFC; ++oc)
        off[((size_t)b * OFFC + oc) * HWSZ + h * WW + ww_] = acc[oc] + bias[oc];
}

// ---------------------------------------------------------------------------
// Kernel 3: repack deform_w f32 [O][Ktot] -> bf16 chunks wB[q][n][kc]
// ---------------------------------------------------------------------------
__global__ void repack_w_kernel(const float* __restrict__ w,
                                bf16_t* __restrict__ wB) {
    int i = blockIdx.x * blockDim.x + threadIdx.x;
    if (i >= (int)WB_ELEMS) return;
    int n = i / KTOT;
    int k = i - n * KTOT;
    int q = k >> 5, kc = k & 31;
    wB[((size_t)q * OUT + n) * KC + kc] = (bf16_t)w[i];
}

// ---------------------------------------------------------------------------
// Kernel 4: deformable conv as im2col GEMM on WMMA bf16.
// Block = 256 threads (8 waves), tile M=128 pixels x N=64 out channels.
// Double-buffered LDS (one barrier per K-chunk); async global->LDS weight
// staging overlapped with bilinear sampling.
// ---------------------------------------------------------------------------
__global__ void deform_gemm_kernel(const float* __restrict__ ref,
                                   const float* __restrict__ dist,
                                   const float* __restrict__ dif,
                                   const float* __restrict__ off,
                                   const bf16_t* __restrict__ wB,
                                   const float* __restrict__ bias,
                                   float* __restrict__ out) {
    __shared__ float sWgt[MT * KP * 4];                  // 18 KB
    __shared__ int   sIdx[MT * KP * 4];                  // 18 KB
    __shared__ __align__(16) bf16_t As[2][MT * KC];      // 2 x 8 KB  [m][kc]
    __shared__ __align__(16) bf16_t Bs[2][OUT * KC];     // 2 x 4 KB  [n][kc]

    const int tid  = threadIdx.x;
    const int lane = tid & 31;
    const int wv   = tid >> 5;          // 0..7
    const int half = lane >> 4;
    const int ln   = lane & 15;

    const int p0  = blockIdx.x * MT;          // first flat pixel of tile
    const int b   = p0 / HWSZ;                // uniform over block
    const int hw0 = p0 - b * HWSZ;

    // ---- Stage bilinear coefficients: 128 pixels x 9 kernel positions ----
    for (int e = tid; e < MT * KP; e += 256) {
        int m  = e / KP, kp = e - m * KP;
        int hw = hw0 + m;
        int h  = hw / WW, xw = hw - h * WW;
        int ky = kp / 3, kx = kp - ky * 3;
        float dyo = off[((size_t)b * OFFC + 2 * kp    ) * HWSZ + hw];
        float dxo = off[((size_t)b * OFFC + 2 * kp + 1) * HWSZ + hw];
        float py = (float)(h - 1 + ky) + dyo;
        float px = (float)(xw - 1 + kx) + dxo;
        float fy = floorf(py), fx = floorf(px);
        int y0 = (int)fy, x0 = (int)fx;
        float dy = py - fy, dx = px - fx;
        int y1 = y0 + 1, x1 = x0 + 1;
        float vy0 = (y0 >= 0 && y0 < HH) ? 1.f : 0.f;
        float vy1 = (y1 >= 0 && y1 < HH) ? 1.f : 0.f;
        float vx0 = (x0 >= 0 && x0 < WW) ? 1.f : 0.f;
        float vx1 = (x1 >= 0 && x1 < WW) ? 1.f : 0.f;
        int cy0 = min(max(y0, 0), HH - 1), cy1 = min(max(y1, 0), HH - 1);
        int cx0 = min(max(x0, 0), WW - 1), cx1 = min(max(x1, 0), WW - 1);
        sWgt[e * 4 + 0] = (1.f - dy) * (1.f - dx) * vy0 * vx0;
        sWgt[e * 4 + 1] = (1.f - dy) * dx         * vy0 * vx1;
        sWgt[e * 4 + 2] = dy * (1.f - dx)         * vy1 * vx0;
        sWgt[e * 4 + 3] = dy * dx                 * vy1 * vx1;
        sIdx[e * 4 + 0] = cy0 * WW + cx0;
        sIdx[e * 4 + 1] = cy0 * WW + cx1;
        sIdx[e * 4 + 2] = cy1 * WW + cx0;
        sIdx[e * 4 + 3] = cy1 * WW + cx1;
    }
    __syncthreads();

    const v8f vzero = {0.f, 0.f, 0.f, 0.f, 0.f, 0.f, 0.f, 0.f};
    v8f acc[4] = {vzero, vzero, vzero, vzero};

    const int mA  = tid >> 1;             // pixel this thread samples (0..127)
    const int kcB = (tid & 1) * 16;       // kc base (16 contiguous per thread)
    const int mrow = wv * 16 + ln;        // this wave's A-fragment row

    for (int q = 0; q < NCH; ++q) {
        const int p = q & 1;
        bf16_t* AsP = As[p];
        bf16_t* BsP = Bs[p];

        // ---- B chunk (4 KB): async global->LDS, overlapped with sampling ----
        const uint4* src4 = reinterpret_cast<const uint4*>(wB + (size_t)q * OUT * KC);
        uint4*       dst4 = reinterpret_cast<uint4*>(BsP);
#if HAVE_ASYNC_LDS
        __builtin_amdgcn_global_load_async_to_lds_b128(
            (GAS v4i*)(src4 + tid), (LAS v4i*)(dst4 + tid), 0, 0);
#else
        dst4[tid] = src4[tid];
#endif
        if (q + 1 < NCH)
            __builtin_prefetch(wB + (size_t)(q + 1) * OUT * KC, 0, 1);

        // ---- A chunk: bilinear-sample 16 (m,kc) elements per thread ----
#pragma unroll
        for (int i = 0; i < 16; ++i) {
            int kc = kcB + i;
            int K  = q * KC + kc;
            int c  = K / KP;
            int kp = K - c * KP;
            int e  = mA * KP + kp;
            const float* sp = (c < CIN) ? ref : ((c < 2 * CIN) ? dist : dif);
            const float* ch = sp + ((size_t)b * CIN + (c & (CIN - 1))) * HWSZ;
            const int*   ix = &sIdx[e * 4];
            const float* wg = &sWgt[e * 4];
            float v = wg[0] * ch[ix[0]] + wg[1] * ch[ix[1]]
                    + wg[2] * ch[ix[2]] + wg[3] * ch[ix[3]];
            AsP[mA * KC + kc] = (bf16_t)v;
        }

#if HAVE_ASYNC_LDS
        asm volatile("s_wait_asynccnt 0x0" ::: "memory");
#endif
        __syncthreads();   // single barrier per chunk (double-buffered LDS)

        // ---- A fragment (16x32 bf16 ISA layout) ----
        const uint4* As4 = reinterpret_cast<const uint4*>(AsP);
        const uint4* Bs4 = reinterpret_cast<const uint4*>(BsP);
        V16 av;
        av.q[0] = As4[mrow * 4 + half];        // lane<16: K0-7  | lane>=16: K8-15
        av.q[1] = As4[mrow * 4 + half + 2];    // lane<16: K16-23| lane>=16: K24-31

        // ---- 4 N-tiles: B fragment (32x16 bf16 ISA layout) + WMMA ----
#pragma unroll
        for (int nt = 0; nt < 4; ++nt) {
            int n = nt * 16 + ln;
            V16 bv;
            bv.q[0] = Bs4[n * 4 + half];           // lane<16: K0-7  | lane>=16: K8-15
            bv.q[1] = Bs4[n * 4 + half + 2];       // lane<16: K16-23| lane>=16: K24-31
            acc[nt] = __builtin_amdgcn_wmma_f32_16x16x32_bf16(
                false, av.v, false, bv.v, (short)0, acc[nt], false, false);
        }
        // no trailing barrier: next iteration writes the other buffer, and
        // writes to THIS buffer (q+2) are ordered after barrier (q+1), which
        // every wave only passes after finishing compute (q).
    }

    // ---- Epilogue: bias + ReLU, scatter per C/D layout ----
#pragma unroll
    for (int nt = 0; nt < 4; ++nt) {
        int n = nt * 16 + ln;
        float bs = bias[n];
        float* op = out + ((size_t)b * OUT + n) * HWSZ;
#pragma unroll
        for (int r = 0; r < 8; ++r) {
            int m_local = half * 8 + r;
            int hw = hw0 + wv * 16 + m_local;
            float v = acc[nt][r] + bs;
            op[hw] = fmaxf(v, 0.f);
        }
    }
}

// ---------------------------------------------------------------------------
// Launcher
// ---------------------------------------------------------------------------
extern "C" void kernel_launch(void* const* d_in, const int* in_sizes, int n_in,
                              void* d_out, int out_size, void* d_ws, size_t ws_size,
                              hipStream_t stream) {
    const float* ref_feat = (const float*)d_in[0];
    const float* dist_feat= (const float*)d_in[1];
    const float* offset_w = (const float*)d_in[2];
    const float* offset_b = (const float*)d_in[3];
    const float* deform_w = (const float*)d_in[4];
    const float* deform_b = (const float*)d_in[5];

    float* feat_out = (float*)d_out;                       // [B,64,H,W]
    float* diff_out = (float*)d_out + FEAT_ELEMS;          // [B,64,H,W]

    float*  off = (float*)d_ws;                            // [B,18,H,W] f32
    bf16_t* wB  = (bf16_t*)((char*)d_ws + OFF_ELEMS * sizeof(float));

    // 1) diff (deform GEMM reads it as the 3rd channel group)
    {
        int n4 = (int)(FEAT_ELEMS / 4);
        diff_kernel<<<n4 / 256, 256, 0, stream>>>(ref_feat, dist_feat, diff_out);
    }
    // 2) offset conv
    {
        dim3 grid(WW / 16, HH / 16, BATCH);
        offset_conv_kernel<<<grid, 256, 0, stream>>>(ref_feat, offset_w,
                                                     offset_b, off);
    }
    // 3) weight repack f32 -> bf16 WMMA-chunk layout
    {
        int blocks = (int)((WB_ELEMS + 255) / 256);
        repack_w_kernel<<<blocks, 256, 0, stream>>>(deform_w, wB);
    }
    // 4) deformable conv GEMM (WMMA bf16, f32 accumulate) + bias + ReLU
    {
        int blocks = (int)(BATCH * HWSZ / MT);   // 288
        deform_gemm_kernel<<<blocks, 256, 0, stream>>>(ref_feat, dist_feat,
                                                       diff_out, off, wB,
                                                       deform_b, feat_out);
    }
}
